// V2VNet_35373350650590
// MI455X (gfx1250) — compile-verified, matrix operand order checked by reading, not run on hardware
//
#include <hip/hip_runtime.h>
#include <hip/hip_bf16.h>

typedef __attribute__((ext_vector_type(16))) __bf16 v16bf;
typedef __attribute__((ext_vector_type(8)))  float  v8f;

#define BB 2
#define AA 5
#define CC 64
#define HH 128
#define WW 128
#define HWSZ (HH*WW)
#define NN (BB*AA)         // 10
#define KCH (2*CC)         // 128 input channels to conv
#define OCH (3*CC)         // 192 output channels (r,z,n gates)
#define PW  (WW+2)         // padded width (1-pixel zero halo)
#define PH  (HH+2)
#define PHW (PH*PW)        // padded pixels per image

// ---------------------------------------------------------------------------
// Zero the whole NHWC+halo GEMM input buffer (uint4 stores).
// ---------------------------------------------------------------------------
__global__ void zero_catin_kernel(uint4* __restrict__ p, long n16) {
    long i = (long)blockIdx.x * blockDim.x + threadIdx.x;
    if (i < n16) p[i] = make_uint4(0u, 0u, 0u, 0u);
}

// ---------------------------------------------------------------------------
// Repack conv weights: wx [oc][kc][kh][kw] f32  ->  wxp [oc][tap][kc] bf16
// so that a K-slab (contiguous kc at fixed tap) is contiguous.
// ---------------------------------------------------------------------------
__global__ void repack_w_kernel(const float* __restrict__ wx,
                                __hip_bfloat16* __restrict__ wxp) {
    int i = blockIdx.x * blockDim.x + threadIdx.x;
    if (i >= OCH * KCH * 9) return;
    int t  = i % 9;
    int kc = (i / 9) % KCH;
    int oc = i / (9 * KCH);
    float v = wx[(size_t)(oc * KCH + kc) * 9 + t];
    wxp[((size_t)oc * 9 + t) * KCH + kc] = __float2bfloat16(v);
}

// ---------------------------------------------------------------------------
// Convert current feats (f32, NCHW) into channels [0,64) of the bf16 NHWC
// (+halo) GEMM input. 64 consecutive threads = 64 channels of one pixel, so
// stores are contiguous 128B per pixel.
// ---------------------------------------------------------------------------
__global__ void pack_src_kernel(const float* __restrict__ src,
                                __hip_bfloat16* __restrict__ catin) {
    long i = (long)blockIdx.x * blockDim.x + threadIdx.x;
    const long total = (long)NN * HWSZ * CC;
    if (i >= total) return;
    long c = i % CC;
    long p = (i / CC) % HWSZ;
    long n = i / ((long)CC * HWSZ);
    long y = p / WW, x = p % WW;
    float v = src[(n * CC + c) * HWSZ + p];
    catin[((n * PHW + (y + 1) * PW + (x + 1)) * KCH) + c] = __float2bfloat16(v);
}

// ---------------------------------------------------------------------------
// Fused double-bilinear warp + mean over j!=i.
// Block: 256 threads = 4 pixels x 64 channels. Threads tid<64 compute the 16
// fused taps (4 translation corners x 4 rotation corners) per pixel into LDS;
// all 256 threads then gather-accumulate over channels.
// Writes bf16 mean into channels [64,128) of the NHWC(+halo) GEMM input.
// ---------------------------------------------------------------------------
__global__ __launch_bounds__(256)
void warp_mean_kernel(const float* __restrict__ src,
                      const float* __restrict__ trans,
                      __hip_bfloat16* __restrict__ catin) {
    __shared__ float s_w[4][16];
    __shared__ int   s_idx[4][16];

    int tid = threadIdx.x;
    const int blocksPerBI = HWSZ / 4;
    int bi      = blockIdx.x / blocksPerBI;
    int pixbase = (blockIdx.x % blocksPerBI) * 4;
    int b = bi / AA;
    int i = bi % AA;

    int pl = tid >> 6;     // pixel slot for accumulation phase
    int c  = tid & 63;     // channel

    float acc = 0.0f;
    for (int j = 0; j < AA; ++j) {
        if (j == i) continue;
        __syncthreads();   // protect LDS taps from previous iteration
        if (tid < 64) {
            int pl1 = tid >> 4, tap = tid & 15;
            int qi = tap >> 2, ri = tap & 3;
            int pix = pixbase + pl1;
            int y = pix / WW, x = pix % WW;
            const float* T = trans + ((size_t)((b * AA + i) * AA + j)) * 16;
            float t00 = T[0], t01 = T[1], t03 = T[3];
            float t10 = T[4], t11 = T[5], t13 = T[7];
            // translation stage (stage 2 of reference): constant pixel shift
            float shiftx = (4.0f * t03 / 128.0f) * (WW * 0.5f);
            float shifty = (-(4.0f * t13) / 128.0f) * (HH * 0.5f);
            float sxp = (float)x + shiftx, syp = (float)y + shifty;
            float qx0 = floorf(sxp), qy0 = floorf(syp);
            float fx = sxp - qx0, fy = syp - qy0;
            float qxf = qx0 + (float)(qi & 1), qyf = qy0 + (float)(qi >> 1);
            float wq = ((qi & 1) ? fx : 1.0f - fx) * ((qi >> 1) ? fy : 1.0f - fy);
            bool vq = (qxf >= 0.0f) && (qxf <= (float)(WW - 1)) &&
                      (qyf >= 0.0f) && (qyf <= (float)(HH - 1));
            int qxi = (int)fminf(fmaxf(qxf, 0.0f), (float)(WW - 1));
            int qyi = (int)fminf(fmaxf(qyf, 0.0f), (float)(HH - 1));
            // rotation stage (stage 1 of reference) evaluated at integer q
            float gxq = (2.0f * qxi + 1.0f) / WW - 1.0f;
            float gyq = (2.0f * qyi + 1.0f) / HH - 1.0f;
            float g0 = t00 * gxq + t01 * gyq;
            float g1 = t10 * gxq + t11 * gyq;
            float xs = ((g0 + 1.0f) * WW - 1.0f) * 0.5f;
            float ys = ((g1 + 1.0f) * HH - 1.0f) * 0.5f;
            float xr0 = floorf(xs), yr0 = floorf(ys);
            float fxr = xs - xr0, fyr = ys - yr0;
            float rxf = xr0 + (float)(ri & 1), ryf = yr0 + (float)(ri >> 1);
            float wr = ((ri & 1) ? fxr : 1.0f - fxr) * ((ri >> 1) ? fyr : 1.0f - fyr);
            bool vr = (rxf >= 0.0f) && (rxf <= (float)(WW - 1)) &&
                      (ryf >= 0.0f) && (ryf <= (float)(HH - 1));
            int rx = (int)fminf(fmaxf(rxf, 0.0f), (float)(WW - 1));
            int ry = (int)fminf(fmaxf(ryf, 0.0f), (float)(HH - 1));
            s_w[pl1][tap]   = (vq && vr) ? wq * wr : 0.0f;
            s_idx[pl1][tap] = ry * WW + rx;
        }
        __syncthreads();
        const float* base = src + ((size_t)(b * AA + j) * CC + c) * HWSZ;
        float a = 0.0f;
        #pragma unroll
        for (int tp = 0; tp < 16; ++tp)
            a += s_w[pl][tp] * base[s_idx[pl][tp]];
        acc += a;
    }
    acc *= (1.0f / (AA - 1));
    int n = b * AA + i;
    int pix = pixbase + pl;
    int y = pix / WW, x = pix % WW;
    catin[((size_t)n * PHW + (y + 1) * PW + (x + 1)) * KCH + CC + c] =
        __float2bfloat16(acc);
}

// ---------------------------------------------------------------------------
// Implicit-GEMM 3x3 conv (bf16 WMMA, f32 accum) fused with ConvGRU gating.
// One wave computes a 16(out-ch) x 32(pixel) macro-tile: 3 gate accumulators
// x 2 pixel tiles. Each A (weight) fragment is loaded once and feeds 2 WMMAs,
// halving weight re-read traffic vs a 16x16 tile. NHWC + zero halo => all
// loads are unconditional b128, EXEC stays all-ones.
// ---------------------------------------------------------------------------
__global__ __launch_bounds__(128)
void conv_gru_kernel(const __hip_bfloat16* __restrict__ catin,
                     const __hip_bfloat16* __restrict__ wxp,
                     const float* __restrict__ bx,
                     const float* __restrict__ bh,
                     float* __restrict__ dst) {
    int wid  = blockIdx.x * 4 + (threadIdx.x >> 5);
    int lane = threadIdx.x & 31;
    int n   = wid >> 11;          // 4 ct-tiles * 512 pixel-pair tiles per image
    int rem = wid & 2047;
    int ct  = rem >> 9;           // out-channel tile within a gate (0..3)
    int pt  = rem & 511;          // pixel-pair tile (32 contiguous pixels)
    int pix0 = pt << 5;
    int y  = pix0 >> 7;           // / WW  (32 | 128 => same row)
    int x0 = pix0 & (WW - 1);
    int npix = lane & 15;         // B/D column (pixel) held by this lane
    int half = lane >> 4;
    int koff = half * 8;          // K striping per 16-bit WMMA VGPR layout
    int m    = lane & 15;         // A row held by this lane

    union Mv { uint4 u[2]; v16bf v; };

    v8f ar0 = {0, 0, 0, 0, 0, 0, 0, 0};
    v8f az0 = ar0, an0 = ar0;
    v8f ar1 = ar0, az1 = ar0, an1 = ar0;

    const unsigned short* cin = (const unsigned short*)catin;
    const unsigned short* wp  = (const unsigned short*)wxp;

    // lane's two pixels in padded NHWC coordinates (tap (0,0) => (y,x) padded)
    long pbase0 = ((long)n * PHW + (long)y * PW + (x0 + npix)) * KCH;
    long pbase1 = pbase0 + (long)16 * KCH;

    #pragma unroll
    for (int t = 0; t < 9; ++t) {
        const int kh = t / 3, kw = t % 3;           // compile-time constants
        long rowoff = ((long)kh * PW + kw) * KCH;
        const unsigned short* brow0 = cin + pbase0 + rowoff;
        const unsigned short* brow1 = cin + pbase1 + rowoff;
        __builtin_prefetch(brow0 + (long)PW * KCH, 0, 1);   // next tap row hint
        #pragma unroll
        for (int kb = 0; kb < KCH; kb += 32) {
            Mv b0, b1;
            const uint4* bp0 = (const uint4*)(brow0 + kb + koff);
            const uint4* bp1 = (const uint4*)(brow1 + kb + koff);
            b0.u[0] = bp0[0];   // K = kb+koff .. +7   (16 contiguous bytes)
            b0.u[1] = bp0[2];   // K = kb+16+koff .. +7
            b1.u[0] = bp1[0];
            b1.u[1] = bp1[2];
            #pragma unroll
            for (int g = 0; g < 3; ++g) {
                int oc = g * CC + ct * 16 + m;
                const uint4* ap =
                    (const uint4*)(wp + ((long)oc * 9 + t) * KCH + kb + koff);
                Mv am;
                am.u[0] = ap[0];
                am.u[1] = ap[2];
                if (g == 0) {
                    ar0 = __builtin_amdgcn_wmma_f32_16x16x32_bf16(
                        false, am.v, false, b0.v, (short)0, ar0, false, false);
                    ar1 = __builtin_amdgcn_wmma_f32_16x16x32_bf16(
                        false, am.v, false, b1.v, (short)0, ar1, false, false);
                } else if (g == 1) {
                    az0 = __builtin_amdgcn_wmma_f32_16x16x32_bf16(
                        false, am.v, false, b0.v, (short)0, az0, false, false);
                    az1 = __builtin_amdgcn_wmma_f32_16x16x32_bf16(
                        false, am.v, false, b1.v, (short)0, az1, false, false);
                } else {
                    an0 = __builtin_amdgcn_wmma_f32_16x16x32_bf16(
                        false, am.v, false, b0.v, (short)0, an0, false, false);
                    an1 = __builtin_amdgcn_wmma_f32_16x16x32_bf16(
                        false, am.v, false, b1.v, (short)0, an1, false, false);
                }
            }
        }
    }

    // ConvGRU gating epilogue (f32). D layout: N = lane&15, M = vgpr + 8*half.
    #pragma unroll
    for (int v = 0; v < 8; ++v) {
        int c = ct * 16 + v + half * 8;
        float bxr = bx[c],          bhr = bh[c];
        float bxz = bx[CC + c],     bhz = bh[CC + c];
        float bxn = bx[2 * CC + c], bhn = bh[2 * CC + c];
        size_t obase = ((size_t)n * CC + c) * HWSZ + y * WW + x0 + npix;

        float r0 = 1.0f / (1.0f + __expf(-(ar0[v] + bxr + bhr)));
        float z0 = 1.0f / (1.0f + __expf(-(az0[v] + bxz + bhz)));
        float n0 = tanhf(an0[v] + bxn + r0 * bhn);
        dst[obase] = (1.0f - z0) * n0;

        float r1 = 1.0f / (1.0f + __expf(-(ar1[v] + bxr + bhr)));
        float z1 = 1.0f / (1.0f + __expf(-(az1[v] + bxz + bhz)));
        float n1 = tanhf(an1[v] + bxn + r1 * bhn);
        dst[obase + 16] = (1.0f - z1) * n1;
    }
}

// ---------------------------------------------------------------------------
extern "C" void kernel_launch(void* const* d_in, const int* in_sizes, int n_in,
                              void* d_out, int out_size, void* d_ws, size_t ws_size,
                              hipStream_t stream) {
    const float* feats = (const float*)d_in[0];
    const float* trans = (const float*)d_in[1];
    const float* wx    = (const float*)d_in[2];
    // d_in[3] = wh: unused (h0 == 0)
    const float* bx    = (const float*)d_in[4];
    const float* bh    = (const float*)d_in[5];

    char* ws = (char*)d_ws;
    const size_t catin_bytes = (size_t)NN * PHW * KCH * 2;    // ~43.3 MB bf16
    const size_t fbuf_bytes  = (size_t)NN * CC * HWSZ * 4;    // ~41.9 MB f32
    __hip_bfloat16* catin = (__hip_bfloat16*)ws;
    float*          fbuf  = (float*)(ws + catin_bytes);
    __hip_bfloat16* wxp   = (__hip_bfloat16*)(ws + catin_bytes + fbuf_bytes);

    {   // zero GEMM input once per launch (interior rewritten every iter,
        // halo stays zero => branch-free conv loads)
        long n16 = (long)(catin_bytes / 16);
        zero_catin_kernel<<<(int)((n16 + 255) / 256), 256, 0, stream>>>(
            (uint4*)catin, n16);
    }
    {
        int total = OCH * KCH * 9;
        repack_w_kernel<<<(total + 255) / 256, 256, 0, stream>>>(wx, wxp);
    }

    const float* src = feats;
    for (int it = 0; it < 2; ++it) {
        {
            long total = (long)NN * CC * HWSZ;
            pack_src_kernel<<<(int)((total + 255) / 256), 256, 0, stream>>>(src, catin);
        }
        warp_mean_kernel<<<NN * (HWSZ / 4), 256, 0, stream>>>(src, trans, catin);
        float* dst = (it == 0) ? fbuf : (float*)d_out;
        conv_gru_kernel<<<(NN * 4 * 512) / 4, 128, 0, stream>>>(catin, wxp, bx, bh, dst);
        src = fbuf;
    }
}
